// TEAttention_82377472737439
// MI455X (gfx1250) — matrix operand order repaired and checked by reading, not compiled, not run on hardware
//
#include <hip/hip_runtime.h>

// ---------------------------------------------------------------------------
// Types / helpers
// ---------------------------------------------------------------------------
typedef __attribute__((ext_vector_type(8)))  float   v8f;
typedef __attribute__((ext_vector_type(8)))  __bf16  bf16x8;
typedef __attribute__((ext_vector_type(16))) __bf16  bf16x16;

#define CAT16(lo, hi) __builtin_shufflevector((lo), (hi), \
    0,1,2,3,4,5,6,7,8,9,10,11,12,13,14,15)
#define WMMA_BF16(a, b, c) __builtin_amdgcn_wmma_f32_16x16x32_bf16( \
    false, (a), false, (b), (short)0, (c), false, false)

constexpr int Bc = 4, Tc = 2048, Cc = 768, Hc = 12, Dc = 64;
constexpr int BT   = Bc * Tc;   // 8192 rows of x
constexpr int NQKV = 3 * Cc;    // 2304

__device__ __forceinline__ unsigned short f2bf(float f) {
    unsigned u = __float_as_uint(f);
    u += 0x7fffu + ((u >> 16) & 1u);           // round-to-nearest-even
    return (unsigned short)(u >> 16);
}

// A-fragment (16x32 bf16, row-major source, rows m0.., leading dim ld)
__device__ __forceinline__ bf16x16 frag_a(const unsigned short* base, int ld,
                                          int m0, int k0, int l15, int lh) {
    const unsigned short* p = base + (m0 + l15) * ld + k0 + lh * 8;
    bf16x8 lo = *(const bf16x8*)(p);
    bf16x8 hi = *(const bf16x8*)(p + 16);
    return CAT16(lo, hi);
}
// B-fragment (32x16 bf16) from W stored row-major (N,K): computes X @ W^T
__device__ __forceinline__ bf16x16 frag_b(const unsigned short* base, int ld,
                                          int n0, int k0, int l15, int lh) {
    const unsigned short* p = base + (n0 + l15) * ld + k0 + lh * 8;
    bf16x8 lo = *(const bf16x8*)(p);
    bf16x8 hi = *(const bf16x8*)(p + 16);
    return CAT16(lo, hi);
}

// ---------------------------------------------------------------------------
// fp32 -> bf16 convert
// ---------------------------------------------------------------------------
__global__ void cvt_bf16_kernel(const float* __restrict__ src,
                                unsigned short* __restrict__ dst, int n) {
    int i = blockIdx.x * blockDim.x + threadIdx.x;
    int stride = gridDim.x * blockDim.x;
    for (; i < n; i += stride) dst[i] = f2bf(src[i]);
}

// ---------------------------------------------------------------------------
// 32Mx64N-per-wave GEMM core, fully unrolled, explicit two-deep ping-pong:
// each buffer set is reloaded IN PLACE right after its last WMMA use, so no
// loop-carried copies are generated and loads run two k-chunks ahead.
// ---------------------------------------------------------------------------
__device__ __forceinline__ void load_ab(const unsigned short* __restrict__ A,
                                        const unsigned short* __restrict__ W,
                                        int lda, int m0, int n0, int k,
                                        int l15, int lh,
                                        bf16x16 a[2], bf16x16 b[4]) {
    a[0] = frag_a(A, lda, m0,      k, l15, lh);
    a[1] = frag_a(A, lda, m0 + 16, k, l15, lh);
    #pragma unroll
    for (int t = 0; t < 4; ++t) b[t] = frag_b(W, lda, n0 + t * 16, k, l15, lh);
}

__device__ __forceinline__ void comp_ab(v8f acc[2][4],
                                        const bf16x16 a[2], const bf16x16 b[4]) {
    #pragma unroll
    for (int t = 0; t < 4; ++t) {
        acc[0][t] = WMMA_BF16(a[0], b[t], acc[0][t]);
        acc[1][t] = WMMA_BF16(a[1], b[t], acc[1][t]);
    }
}

__device__ __forceinline__ void gemm_32x64(const unsigned short* __restrict__ A,
                                           const unsigned short* __restrict__ W,
                                           int lda, int m0, int n0,
                                           int l15, int lh, v8f acc[2][4]) {
    bf16x16 a0[2], b0[4], a1[2], b1[4];
    load_ab(A, W, lda, m0, n0, 0,  l15, lh, a0, b0);
    load_ab(A, W, lda, m0, n0, 32, l15, lh, a1, b1);
    #pragma unroll
    for (int k = 0; k < Cc; k += 64) {
        comp_ab(acc, a0, b0);
        if (k + 64 < Cc) load_ab(A, W, lda, m0, n0, k + 64, l15, lh, a0, b0);
        comp_ab(acc, a1, b1);
        if (k + 96 < Cc) load_ab(A, W, lda, m0, n0, k + 96, l15, lh, a1, b1);
    }
}

// ---------------------------------------------------------------------------
// QKV projection (8192x2304 = x @ w_qkv^T) + RoPE epilogue + scatter
//   Q -> (B,H,T,D) bf16 (scaled by 1/sqrt(D)), K -> (B,H,T,D), V -> (B,H,D,T)
// block = 128 threads (4 waves), tile = 128(M) x 64(N), wave = 32(M) x 64(N)
// ---------------------------------------------------------------------------
__global__ void __launch_bounds__(128)
qkv_rope_kernel(const unsigned short* __restrict__ xb,
                const unsigned short* __restrict__ wb,
                const float* __restrict__ rope,      // (T, D/2, 2) fp32
                unsigned short* __restrict__ Qb,
                unsigned short* __restrict__ Kb,
                unsigned short* __restrict__ Vt) {
    const int lane = threadIdx.x & 31;
    const int wid  = threadIdx.x >> 5;
    const int l15  = lane & 15;
    const int lh   = lane >> 4;
    const int m0   = blockIdx.x * 128 + wid * 32;
    const int n0   = blockIdx.y * 64;

    v8f acc[2][4] = {};
    gemm_32x64(xb, wb, Cc, m0, n0, l15, lh, acc);

    // epilogue: RoPE + scatter. C-layout: elem(mt,t,v) at row m0+mt*16+v+8*lh,
    // col n0+t*16+l15. Pair partner for RoPE sits on lane^1.
    #pragma unroll
    for (int mt = 0; mt < 2; ++mt)
        #pragma unroll
        for (int t = 0; t < 4; ++t) {
            const int col   = n0 + t * 16 + l15;
            const int which = col / Cc;          // 0=q 1=k 2=v (wave-uniform)
            const int r     = col % Cc;
            const int h     = r / Dc;
            const int d     = r % Dc;
            #pragma unroll
            for (int v = 0; v < 8; ++v) {
                const int row = m0 + mt * 16 + v + 8 * lh;   // 0..8191
                const int b   = row >> 11;                   // /T
                const int tt  = row & (Tc - 1);              // %T
                const int bh  = b * Hc + h;
                float val     = acc[mt][t][v];
                float partner = __shfl_xor(val, 1, 32);
                if (which == 2) {
                    Vt[(bh * Dc + d) * Tc + tt] = f2bf(val);
                } else {
                    const float2 cs =
                        ((const float2*)rope)[tt * (Dc / 2) + (d >> 1)];
                    float o = ((d & 1) == 0) ? (val * cs.x - partner * cs.y)
                                             : (partner * cs.y + val * cs.x);
                    if (which == 0) {
                        Qb[(bh * Tc + tt) * Dc + d] = f2bf(o * 0.125f);
                    } else {
                        Kb[(bh * Tc + tt) * Dc + d] = f2bf(o);
                    }
                }
            }
        }
}

// ---------------------------------------------------------------------------
// Flash attention: per wave 16 query rows, stream 32-key chunks (causal).
// block = 128 threads (4 waves), grid = (T/64, B*H). No barriers (waves have
// different causal trip counts); P C->A relayout via wave-private LDS tile.
// V fragments are loaded before the softmax update so their latency overlaps
// the VALU-heavy online-softmax work. Output: (B,T,H*D) bf16 row-major.
// ---------------------------------------------------------------------------
__global__ void __launch_bounds__(128)
attn_kernel(const unsigned short* __restrict__ Qb,
            const unsigned short* __restrict__ Kb,
            const unsigned short* __restrict__ Vt,
            unsigned short* __restrict__ attn) {
    __shared__ __align__(16) unsigned short plds[4][16][32];
    const int lane = threadIdx.x & 31;
    const int wid  = threadIdx.x >> 5;
    const int l15  = lane & 15;
    const int lh   = lane >> 4;
    const int bh   = blockIdx.y;
    const int m0   = blockIdx.x * 64 + wid * 16;

    const unsigned short* q  = Qb + bh * Tc * Dc;
    const unsigned short* kk = Kb + bh * Tc * Dc;
    const unsigned short* vt = Vt + bh * Tc * Dc;   // (D,T) layout

    bf16x16 qa[2];
    #pragma unroll
    for (int kc = 0; kc < 2; ++kc) qa[kc] = frag_a(q, Dc, m0, kc * 32, l15, lh);

    v8f o[4] = {};
    float mrun[8], lrun[8];
    #pragma unroll
    for (int v = 0; v < 8; ++v) { mrun[v] = -1e30f; lrun[v] = 0.0f; }

    const int nch = (m0 + 16 + 31) >> 5;            // causal chunk count
    for (int j = 0; j < nch; ++j) {
        const int kb = j * 32;
        v8f s[2] = {};
        #pragma unroll
        for (int t = 0; t < 2; ++t) {
            const int kbt = kb + t * 16;
            #pragma unroll
            for (int kc = 0; kc < 2; ++kc) {
                bf16x16 bfr = frag_b(kk, Dc, kbt, kc * 32, l15, lh);
                s[t] = WMMA_BF16(qa[kc], bfr, s[t]);
            }
        }
        // issue V-fragment loads now; consumed only after the softmax update
        bf16x16 vb[4];
        #pragma unroll
        for (int t = 0; t < 4; ++t) vb[t] = frag_b(vt, Tc, t * 16, kb, l15, lh);

        // causal mask on diagonal chunk
        #pragma unroll
        for (int t = 0; t < 2; ++t) {
            const int kbt = kb + t * 16;
            if (kbt + 15 > m0) {
                const int key = kbt + l15;
                #pragma unroll
                for (int v = 0; v < 8; ++v)
                    if (key > m0 + v + 8 * lh) s[t][v] = -1e30f;
            }
        }
        // online softmax: row stats via 16-lane xor reductions
        float alpha[8], mnew[8];
        #pragma unroll
        for (int v = 0; v < 8; ++v) {
            float x = fmaxf(s[0][v], s[1][v]);
            x = fmaxf(x, __shfl_xor(x, 1, 32));
            x = fmaxf(x, __shfl_xor(x, 2, 32));
            x = fmaxf(x, __shfl_xor(x, 4, 32));
            x = fmaxf(x, __shfl_xor(x, 8, 32));
            float mn = fmaxf(mrun[v], x);
            mnew[v]  = mn;
            alpha[v] = __expf(mrun[v] - mn);
            mrun[v]  = mn;
        }
        #pragma unroll
        for (int v = 0; v < 8; ++v) {
            float p0 = __expf(s[0][v] - mnew[v]);
            float p1 = __expf(s[1][v] - mnew[v]);
            s[0][v] = p0; s[1][v] = p1;
            float rs = p0 + p1;
            rs += __shfl_xor(rs, 1, 32);
            rs += __shfl_xor(rs, 2, 32);
            rs += __shfl_xor(rs, 4, 32);
            rs += __shfl_xor(rs, 8, 32);
            lrun[v] = lrun[v] * alpha[v] + rs;
        }
        // P (C-layout) -> LDS -> A-fragment, rescale O meanwhile
        #pragma unroll
        for (int t = 0; t < 2; ++t)
            #pragma unroll
            for (int v = 0; v < 8; ++v)
                plds[wid][v + 8 * lh][t * 16 + l15] = f2bf(s[t][v]);
        #pragma unroll
        for (int t = 0; t < 4; ++t)
            #pragma unroll
            for (int v = 0; v < 8; ++v) o[t][v] *= alpha[v];

        bf16x8 plo = *(const bf16x8*)&plds[wid][l15][lh * 8];
        bf16x8 phi = *(const bf16x8*)&plds[wid][l15][16 + lh * 8];
        bf16x16 pa = CAT16(plo, phi);

        #pragma unroll
        for (int t = 0; t < 4; ++t) o[t] = WMMA_BF16(pa, vb[t], o[t]);
    }

    // epilogue: normalize, scatter to (B,T,C) bf16
    const int b = bh / Hc, h = bh % Hc;
    #pragma unroll
    for (int t = 0; t < 4; ++t)
        #pragma unroll
        for (int v = 0; v < 8; ++v) {
            const int row = m0 + v + 8 * lh;
            const int d   = t * 16 + l15;
            attn[(b * Tc + row) * Cc + h * Dc + d] = f2bf(o[t][v] / lrun[v]);
        }
}

// ---------------------------------------------------------------------------
// Output projection: (8192x768) = attn @ w_o^T, fp32 out
// block = 128 threads (4 waves), tile = 128(M) x 64(N), wave = 32(M) x 64(N)
// ---------------------------------------------------------------------------
__global__ void __launch_bounds__(128)
oproj_kernel(const unsigned short* __restrict__ attn,
             const unsigned short* __restrict__ wob,
             float* __restrict__ out) {
    const int lane = threadIdx.x & 31;
    const int wid  = threadIdx.x >> 5;
    const int l15  = lane & 15;
    const int lh   = lane >> 4;
    const int m0   = blockIdx.x * 128 + wid * 32;
    const int n0   = blockIdx.y * 64;

    v8f acc[2][4] = {};
    gemm_32x64(attn, wob, Cc, m0, n0, l15, lh, acc);

    #pragma unroll
    for (int mt = 0; mt < 2; ++mt)
        #pragma unroll
        for (int t = 0; t < 4; ++t)
            #pragma unroll
            for (int v = 0; v < 8; ++v)
                out[(m0 + mt * 16 + v + 8 * lh) * Cc + n0 + t * 16 + l15] =
                    acc[mt][t][v];
}

// ---------------------------------------------------------------------------
// Launch
// ---------------------------------------------------------------------------
extern "C" void kernel_launch(void* const* d_in, const int* in_sizes, int n_in,
                              void* d_out, int out_size, void* d_ws, size_t ws_size,
                              hipStream_t stream) {
    const float* x    = (const float*)d_in[0];   // (B,T,C)
    const float* rope = (const float*)d_in[1];   // (T,D/2,2)
    const float* wqkv = (const float*)d_in[2];   // (3C,C)
    const float* wo   = (const float*)d_in[3];   // (C,C)
    float* out = (float*)d_out;

    // bf16 workspace layout (ushort units)
    unsigned short* ws = (unsigned short*)d_ws;
    constexpr size_t SZ_X  = (size_t)BT * Cc;           // 6291456
    constexpr size_t SZ_WQ = (size_t)NQKV * Cc;         // 1769472
    constexpr size_t SZ_WO = (size_t)Cc * Cc;           // 589824
    constexpr size_t SZ_H  = (size_t)Bc * Hc * Tc * Dc; // 6291456
    unsigned short* xb   = ws;
    unsigned short* wqb  = xb  + SZ_X;
    unsigned short* wob  = wqb + SZ_WQ;
    unsigned short* Qb   = wob + SZ_WO;
    unsigned short* Kb   = Qb  + SZ_H;
    unsigned short* Vt   = Kb  + SZ_H;
    unsigned short* attn = Vt  + SZ_H;                  // total ~67.6 MB

    cvt_bf16_kernel<<<2048, 256, 0, stream>>>(x, xb, (int)SZ_X);
    cvt_bf16_kernel<<<2048, 256, 0, stream>>>(wqkv, wqb, (int)SZ_WQ);
    cvt_bf16_kernel<<<1024, 256, 0, stream>>>(wo, wob, (int)SZ_WO);

    qkv_rope_kernel<<<dim3(BT / 128, NQKV / 64), 128, 0, stream>>>(
        xb, wqb, rope, Qb, Kb, Vt);

    attn_kernel<<<dim3(Tc / 64, Bc * Hc), 128, 0, stream>>>(Qb, Kb, Vt, attn);

    oproj_kernel<<<dim3(BT / 128, Cc / 64), 128, 0, stream>>>(attn, wob, out);
}
